// DynamicRNNV2_11879879543236
// MI455X (gfx1250) — matrix operand order repaired and checked by reading, not compiled, not run on hardware
//
#include <hip/hip_runtime.h>
#include <hip/hip_bf16.h>
#include <stdint.h>

// Problem constants (match reference)
#define B_   32
#define T_   1024
#define D_   1024
#define H_   1024
#define N3_  3072
#define M_   (B_ * T_)   // 32768 rows for the x-projection GEMM

typedef __attribute__((ext_vector_type(16))) __bf16 v16bf;
typedef __attribute__((ext_vector_type(8)))  float  v8f;
typedef __attribute__((ext_vector_type(4)))  unsigned int v4u;
typedef __attribute__((ext_vector_type(8)))  int v8i;
typedef __attribute__((ext_vector_type(4)))  int v4i;

#if defined(__has_builtin)
#if __has_builtin(__builtin_amdgcn_tensor_load_to_lds) && \
    __has_builtin(__builtin_amdgcn_s_wait_tensorcnt)
#define USE_TDM 1
#else
#define USE_TDM 0
#endif
#else
#define USE_TDM 0
#endif

union Frag16 {          // 32 bytes: one WMMA bf16 A or B operand
  v16bf v;
  uint4 q[2];
};

__device__ __forceinline__ uint32_t bfbits(float f) {
  uint32_t u = __builtin_bit_cast(uint32_t, f);
  u += 0x7FFFu + ((u >> 16) & 1u);             // round-to-nearest-even
  return u >> 16;
}
__device__ __forceinline__ uint32_t pack2(float a, float b) {
  return bfbits(a) | (bfbits(b) << 16);
}
__device__ __forceinline__ __bf16 f2bf(float f) {
  uint16_t s = (uint16_t)bfbits(f);
  return __builtin_bit_cast(__bf16, s);
}

// ---------------------------------------------------------------------------
// Kernel 0: input_data fp32 -> bf16 (packed, coalesced). 64 MB; L2-resident.
// ---------------------------------------------------------------------------
__global__ __launch_bounds__(256)
void cvt_x(const float* __restrict__ X, __bf16* __restrict__ Xb) {
  int i = blockIdx.x * 256 + threadIdx.x;      // float4 index
  float4 f = ((const float4*)X)[i];
  uint2 p;
  p.x = pack2(f.x, f.y);
  p.y = pack2(f.z, f.w);
  ((uint2*)Xb)[i] = p;
}

// ---------------------------------------------------------------------------
// Kernel 1: transpose + convert [K=1024][N=3072] fp32 -> bf16 [N=3072][K=1024]
// Used for both `kernel` (-> WT) and `recurrent_kernel` (-> recT).
// ---------------------------------------------------------------------------
__global__ __launch_bounds__(256)
void transpose_cvt(const float* __restrict__ U, __bf16* __restrict__ VT) {
  __shared__ float tile[32][33];
  const int tx = threadIdx.x;            // 0..31
  const int ty = threadIdx.y;            // 0..7
  const int n0 = blockIdx.x * 32;
  const int k0 = blockIdx.y * 32;
#pragma unroll
  for (int i = 0; i < 4; ++i)
    tile[ty + 8 * i][tx] = U[(size_t)(k0 + ty + 8 * i) * N3_ + n0 + tx];
  __syncthreads();
#pragma unroll
  for (int i = 0; i < 4; ++i) {
    int n = ty + 8 * i;
    VT[(size_t)(n0 + n) * H_ + k0 + tx] = f2bf(tile[tx][n]);
  }
}

// ---------------------------------------------------------------------------
// TDM: DMA one 128-row x 64-col bf16 tile of Xb[., k0..k0+64) into LDS with
// 16B/row padding (LDS row stride 144B). Issued by one wave; EXEC ignored.
// D# packing per CDNA5 ISA.
// This toolchain exposes the 6-arg builtin:
//   (uint32x4 g0, int32x8 g1, int32x4 g2, int32x4 g3, int32x8, i32 cpol)
// ---------------------------------------------------------------------------
#if USE_TDM
__device__ __forceinline__ void tdm_load_tile(const __bf16* gsrc,
                                              uint32_t lds_off) {
  uint64_t ga = (uint64_t)(uintptr_t)gsrc;
  v4u g0;
  g0.x = 1u;                                        // count=1, user mode
  g0.y = lds_off;                                   // lds_addr (bytes)
  g0.z = (uint32_t)ga;                              // global_addr[31:0]
  g0.w = (uint32_t)((ga >> 32) & 0x01FFFFFFu)       // global_addr[56:32]
         | 0x80000000u;                             // type=2 ("image")
  v8i g1;
  g1[0] = (int)((1u << 16)      // data_size=1 -> 2 bytes
              | (1u << 20)      // pad_enable
              | (4u << 22)      // pad_interval code 4 -> every 32 DWORDs (128B)
              | (3u << 25));    // pad_amount  code 3 -> 4 DWORDs (16B)
  g1[1] = (int)(1024u << 16);   // tensor_dim0[15:0]=1024 (bits 63:48)
  g1[2] = (int)0x80000000u;     // tensor_dim1[15:0]=32768 (bits 95:80)
  g1[3] = (int)(64u << 16);     // tile_dim0=64 (bits 127:112)
  g1[4] = (int)128u;            // tile_dim1=128 (bits 143:128), tile_dim2=0
  g1[5] = (int)1024u;           // tensor_dim0_stride[31:0]=1024 elements
  g1[6] = 0;
  g1[7] = 0;
  v4i z4 = {0, 0, 0, 0};                   // groups 2/3 unused (2D tile)
  v8i z8 = {0, 0, 0, 0, 0, 0, 0, 0};
  __builtin_amdgcn_tensor_load_to_lds(g0, g1, z4, z4, z8, 0);
}
#endif

// ---------------------------------------------------------------------------
// Kernel 2: x_proj = input @ kernel + b_in   (M=32768, K=1024, N=3072)
// 256 threads = 8 waves, 128x128 tile, BK=64.
// A: bf16 tile DMA'd (TDM) into LDS, double-buffered; frags via ds_load_b128.
// B: fragments loaded straight from L2-resident pre-transposed WT (global).
// ---------------------------------------------------------------------------
__global__ __launch_bounds__(256)
void xproj_gemm(const __bf16* __restrict__ Xb, const __bf16* __restrict__ WT,
                const float* __restrict__ b_in, float* __restrict__ P) {
  __shared__ __bf16 As[2][128][72];   // 64 data + 8 pad bf16 (144B row stride)

  const int tid  = threadIdx.x;
  const int lane = tid & 31;
  const int w    = tid >> 5;      // wave 0..7
  const int wm   = w >> 1;        // 0..3 : 32-row slab
  const int wn   = w & 1;         // 0..1 : 64-col slab
  const int l16  = lane & 15;
  const int hi   = lane >> 4;     // lane half
  const int kA   = hi * 16;       // A: K chunks {0-15} / {16-31}
  const int kB   = hi * 16;       // B: K chunks {0-15} / {16-31}

  const int m0 = blockIdx.y * 128;
  const int n0 = blockIdx.x * 128;
  const int NT = D_ / 64;         // 16 K-tiles

  // B row pointers (columns of original kernel), L2-resident
  const __bf16* brow[4];
#pragma unroll
  for (int fn = 0; fn < 4; ++fn)
    brow[fn] = WT + (size_t)(n0 + wn * 64 + fn * 16 + l16) * D_;

  v8f acc[2][4] = {};

#if USE_TDM
  if (w == 0)
    tdm_load_tile(Xb + (size_t)m0 * D_, (uint32_t)(uintptr_t)&As[0][0][0]);
#endif

  for (int it = 0; it < NT; ++it) {
    const int cur = it & 1;
    const int k0  = it * 64;

#if USE_TDM
    if (it + 1 < NT) {
      if (w == 0) {
        tdm_load_tile(Xb + (size_t)m0 * D_ + (k0 + 64),
                      (uint32_t)(uintptr_t)&As[1 - cur][0][0]);
        __builtin_amdgcn_s_wait_tensorcnt(1);   // current tile complete
      }
    } else {
      if (w == 0) __builtin_amdgcn_s_wait_tensorcnt(0);
    }
    __syncthreads();                            // tile visible to all waves
#else
    // fallback: vectorized copy staging (Xb already bf16)
#pragma unroll
    for (int i = 0; i < 4; ++i) {
      int vec = tid + i * 256;                  // 1024 x 16B chunks
      int r   = vec >> 3;
      int c8  = (vec & 7) * 8;
      *(uint4*)&As[cur][r][c8] =
          *(const uint4*)(Xb + (size_t)(m0 + r) * D_ + k0 + c8);
    }
    __syncthreads();
#endif

    // two K=32 WMMA steps per staged tile
#pragma unroll
    for (int kk = 0; kk < 64; kk += 32) {
      Frag16 a[2], b[4];
#pragma unroll
      for (int fm = 0; fm < 2; ++fm) {
        int row = wm * 32 + fm * 16 + l16;
        a[fm].q[0] = *(const uint4*)&As[cur][row][kk + kA];
        a[fm].q[1] = *(const uint4*)&As[cur][row][kk + kA + 8];
      }
#pragma unroll
      for (int fn = 0; fn < 4; ++fn) {
        b[fn].q[0] = *(const uint4*)(brow[fn] + k0 + kk + kB);
        b[fn].q[1] = *(const uint4*)(brow[fn] + k0 + kk + kB + 8);
      }
#pragma unroll
      for (int fm = 0; fm < 2; ++fm)
#pragma unroll
        for (int fn = 0; fn < 4; ++fn)
          acc[fm][fn] = __builtin_amdgcn_wmma_f32_16x16x32_bf16(
              false, a[fm].v, false, b[fn].v, (short)0, acc[fm][fn],
              false, false);
    }
    __syncthreads();   // all reads of As[cur] done before it is overwritten
  }

  // epilogue: + input bias, store fp32
#pragma unroll
  for (int fm = 0; fm < 2; ++fm) {
#pragma unroll
    for (int fn = 0; fn < 4; ++fn) {
      int   n_g  = n0 + wn * 64 + fn * 16 + l16;
      float bias = b_in[n_g];
#pragma unroll
      for (int r = 0; r < 8; ++r) {
        int m_g = m0 + wm * 32 + fm * 16 + r + hi * 8;   // VGPR r -> row r/r+8
        P[(size_t)m_g * N3_ + n_g] = acc[fm][fn][r] + bias;
      }
    }
  }
}

// ---------------------------------------------------------------------------
// Device-wide barrier (persistent scan kernel; 64 blocks are co-resident)
// ---------------------------------------------------------------------------
__device__ __forceinline__ void grid_barrier(unsigned* cnt, unsigned* gen,
                                             unsigned nblocks) {
  __syncthreads();
  if (threadIdx.x == 0) {
    unsigned g = __hip_atomic_load(gen, __ATOMIC_ACQUIRE, __HIP_MEMORY_SCOPE_AGENT);
    unsigned arrived =
        __hip_atomic_fetch_add(cnt, 1u, __ATOMIC_ACQ_REL, __HIP_MEMORY_SCOPE_AGENT);
    if (arrived == nblocks - 1u) {
      __hip_atomic_store(cnt, 0u, __ATOMIC_RELAXED, __HIP_MEMORY_SCOPE_AGENT);
      __hip_atomic_fetch_add(gen, 1u, __ATOMIC_ACQ_REL, __HIP_MEMORY_SCOPE_AGENT);
    } else {
      while (__hip_atomic_load(gen, __ATOMIC_ACQUIRE, __HIP_MEMORY_SCOPE_AGENT) == g)
        __builtin_amdgcn_s_sleep(8);
    }
  }
  __syncthreads();
}

// ---------------------------------------------------------------------------
// Kernel 3: persistent GRU scan. 64 blocks x 64 threads (2 waves).
// Block owns 16 hidden columns j0..j0+15; computes gate cols j, H+j, 2H+j.
// B frags from L2-resident recT; A frags from ping-pong bf16 h buffer.
// One grid barrier per step.
// ---------------------------------------------------------------------------
__global__ __launch_bounds__(64)
void gru_scan(const float* __restrict__ P, const __bf16* __restrict__ recT,
              const float* __restrict__ b_rec, float* __restrict__ out,
              float* __restrict__ h_f32, __bf16* __restrict__ hb0,
              __bf16* __restrict__ hb1, unsigned* bar_cnt, unsigned* bar_gen) {
  const int tid  = threadIdx.x;
  const int lane = tid & 31;
  const int w    = tid >> 5;      // wave: rows 0-15 / 16-31 of h
  const int l16  = lane & 15;
  const int hi   = lane >> 4;
  const int kA   = hi * 16;
  const int kB   = hi * 16;
  const int j0   = blockIdx.x * 16;
  const unsigned nb = gridDim.x;

  // zero-init this block's state slice (both ping-pong buffers + f32 state)
  for (int idx = tid; idx < 512; idx += 64) {
    int m = idx >> 4, jj = idx & 15;
    h_f32[m * H_ + j0 + jj] = 0.0f;
    hb0[m * H_ + j0 + jj] = f2bf(0.0f);
    hb1[m * H_ + j0 + jj] = f2bf(0.0f);
  }
  __threadfence();
  grid_barrier(bar_cnt, bar_gen, nb);

  const int mrow = w * 16 + l16;                         // A-matrix row
  const __bf16* bz_row = recT + (size_t)(0 * H_ + j0 + l16) * H_;
  const __bf16* br_row = recT + (size_t)(1 * H_ + j0 + l16) * H_;
  const __bf16* bh_row = recT + (size_t)(2 * H_ + j0 + l16) * H_;

  for (int t = 0; t < T_; ++t) {
    const __bf16* hcur = (t & 1) ? hb1 : hb0;
    __bf16*       hnxt = (t & 1) ? hb0 : hb1;
    const __bf16* arow = hcur + (size_t)mrow * H_;

    v8f accz = {}, accr = {}, acch = {};
    for (int k = 0; k < H_; k += 32) {
      Frag16 a, bz, br, bh;
      a.q[0]  = *(const uint4*)(arow + k + kA);
      a.q[1]  = *(const uint4*)(arow + k + kA + 8);
      bz.q[0] = *(const uint4*)(bz_row + k + kB);
      bz.q[1] = *(const uint4*)(bz_row + k + kB + 8);
      br.q[0] = *(const uint4*)(br_row + k + kB);
      br.q[1] = *(const uint4*)(br_row + k + kB + 8);
      bh.q[0] = *(const uint4*)(bh_row + k + kB);
      bh.q[1] = *(const uint4*)(bh_row + k + kB + 8);
      accz = __builtin_amdgcn_wmma_f32_16x16x32_bf16(false, a.v, false, bz.v,
                                                     (short)0, accz, false, false);
      accr = __builtin_amdgcn_wmma_f32_16x16x32_bf16(false, a.v, false, br.v,
                                                     (short)0, accr, false, false);
      acch = __builtin_amdgcn_wmma_f32_16x16x32_bf16(false, a.v, false, bh.v,
                                                     (short)0, acch, false, false);
    }

    // gate nonlinearity + state update (each (m,j) owned by one lane)
#pragma unroll
    for (int r = 0; r < 8; ++r) {
      int m = w * 16 + r + hi * 8;
      int j = j0 + l16;
      size_t xrow = ((size_t)m * T_ + t) * N3_;
      float rz = accz[r] + b_rec[j];
      float rr = accr[r] + b_rec[H_ + j];
      float rh = acch[r] + b_rec[2 * H_ + j];
      float xz = P[xrow + j];
      float xr = P[xrow + H_ + j];
      float xh = P[xrow + 2 * H_ + j];
      float z  = 1.0f / (1.0f + __expf(-(xz + rz)));
      float rg = 1.0f / (1.0f + __expf(-(xr + rr)));
      float hh = tanhf(xh + rg * rh);
      float ho = h_f32[m * H_ + j];
      float hn = z * ho + (1.0f - z) * hh;
      out[((size_t)m * T_ + t) * H_ + j] = hn;
      h_f32[m * H_ + j] = hn;
      hnxt[m * H_ + j]  = f2bf(hn);
      if (t == T_ - 1)
        out[(size_t)B_ * T_ * H_ + (size_t)m * H_ + j] = hn;   // h_last
    }
    __threadfence();
    grid_barrier(bar_cnt, bar_gen, nb);
  }
}

// ---------------------------------------------------------------------------
// Workspace layout (bytes, all 256B-aligned)
// ---------------------------------------------------------------------------
#define OFF_P     ((size_t)0)                                   // 384 MB fp32
#define OFF_RECT  (OFF_P    + (size_t)M_ * N3_ * 4)             // 6 MB bf16
#define OFF_WT    (OFF_RECT + (size_t)N3_ * H_ * 2)             // 6 MB bf16
#define OFF_XB    (OFF_WT   + (size_t)N3_ * H_ * 2)             // 64 MB bf16
#define OFF_H32   (OFF_XB   + (size_t)M_ * D_ * 2)              // 128 KB
#define OFF_HB0   (OFF_H32  + (size_t)B_ * H_ * 4)              // 64 KB
#define OFF_HB1   (OFF_HB0  + (size_t)B_ * H_ * 2)              // 64 KB
#define OFF_BAR   (OFF_HB1  + (size_t)B_ * H_ * 2)              // 256 B

extern "C" void kernel_launch(void* const* d_in, const int* in_sizes, int n_in,
                              void* d_out, int out_size, void* d_ws, size_t ws_size,
                              hipStream_t stream) {
  const float* X    = (const float*)d_in[0];   // [B,T,D]
  const float* W    = (const float*)d_in[1];   // [D,3H]
  const float* U    = (const float*)d_in[2];   // [H,3H]
  const float* bias = (const float*)d_in[3];   // [2,3H]
  float* out = (float*)d_out;

  char* ws = (char*)d_ws;
  float*    P    = (float*)(ws + OFF_P);
  __bf16*   recT = (__bf16*)(ws + OFF_RECT);
  __bf16*   WT   = (__bf16*)(ws + OFF_WT);
  __bf16*   Xb   = (__bf16*)(ws + OFF_XB);
  float*    h32  = (float*)(ws + OFF_H32);
  __bf16*   hb0  = (__bf16*)(ws + OFF_HB0);
  __bf16*   hb1  = (__bf16*)(ws + OFF_HB1);
  unsigned* bar  = (unsigned*)(ws + OFF_BAR);

  // deterministic barrier state every call (capturable)
  (void)hipMemsetAsync(bar, 0, 256, stream);

  cvt_x<<<dim3((M_ * D_ / 4) / 256), 256, 0, stream>>>(X, Xb);

  transpose_cvt<<<dim3(N3_ / 32, H_ / 32), dim3(32, 8), 0, stream>>>(W, WT);
  transpose_cvt<<<dim3(N3_ / 32, H_ / 32), dim3(32, 8), 0, stream>>>(U, recT);

  xproj_gemm<<<dim3(N3_ / 128, M_ / 128), 256, 0, stream>>>(Xb, WT, bias, P);

  gru_scan<<<dim3(H_ / 16), 64, 0, stream>>>(P, recT, bias + N3_, out,
                                             h32, hb0, hb1, bar, bar + 64);
}
